// TopKastLinear_30279519437502
// MI455X (gfx1250) — matrix operand order repaired and checked by reading, not compiled, not run on hardware
//
#include <hip/hip_runtime.h>

typedef unsigned int       u32;
typedef unsigned short     u16;
typedef unsigned long long u64;
typedef __attribute__((ext_vector_type(4)))  u32    v4u;
typedef __attribute__((ext_vector_type(2)))  u32    v2u;
typedef __attribute__((ext_vector_type(8)))  int    v8i;
typedef __attribute__((ext_vector_type(4)))  int    v4i;
typedef __attribute__((ext_vector_type(16))) __bf16 v16bf;
typedef __attribute__((ext_vector_type(8)))  float  v8f;

#define B_    1024
#define IN_   4096
#define OUT_  4096
#define NW    (OUT_ * IN_)      // 16777216 weights
#define RANK0 8388607u          // quantile(0.5) interpolates sorted[8388607], sorted[8388608]

// ---- order-preserving float<->uint key ----
__device__ __forceinline__ u32 fkey(u32 u) {
  u32 m = (u32)((int)u >> 31) | 0x80000000u;
  return u ^ m;
}
__device__ __forceinline__ float keyval(u32 k) {
  u32 u = (k & 0x80000000u) ? (k ^ 0x80000000u) : ~k;
  return __uint_as_float(u);
}
__device__ __forceinline__ u16 f2bf(float x) {  // round-to-nearest-even
  u32 u = __float_as_uint(x);
  return (u16)((u + 0x7FFFu + ((u >> 16) & 1u)) >> 16);
}
__device__ __forceinline__ float bf2f(u16 h) {
  return __uint_as_float(((u32)h) << 16);
}

// ---- CDNA5 TDM (Tensor Data Mover) helpers ----
#if __has_builtin(__builtin_amdgcn_tensor_load_to_lds)
#define HAVE_TDM 1
// 2D bf16 tile load: tile_dim0 = 32 elems (64B row), tile_dim1 = rows,
// tensor_dim0_stride = IN_ (row-major source), LDS pad 16B after every 64B
// row so destination rows land on an 80B (40-half) stride.
// This toolchain exposes the 6-arg builtin:
//   (uint32x4 g0, int32x8 g1, int32x4 g2, int32x4 g3, int32x8 g4, i32 cpol)
__device__ __forceinline__ void tdm_load_tile(u32 lds_addr, const u16* gsrc, int rows) {
  u64 ga = (u64)(uintptr_t)gsrc;
  v4u g0;
  g0[0] = 1u;                                  // count=1 valid user descriptor
  g0[1] = lds_addr;                            // LDS byte address of tile start
  g0[2] = (u32)ga;                             // global_addr[31:0]
  g0[3] = (u32)(ga >> 32) | (2u << 30);        // global_addr[56:32] | type=2 ("image")
  v8i g1;
  g1[0] = (int)((1u << 16)                     // data_size = 2 bytes (bf16)
              | (1u << 20)                     // pad_enable
              | (3u << 22)                     // pad_interval: 16 DWORDs = 64B
              | (3u << 25));                   // pad_amount:   4 DWORDs = 16B
  g1[1] = (int)(((u32)IN_ & 0xFFFFu) << 16);                                   // tensor_dim0[15:0]
  g1[2] = (int)((((u32)IN_ >> 16) & 0xFFFFu) | (((u32)IN_ & 0xFFFFu) << 16));  // dim0 hi | dim1 lo
  g1[3] = (int)((((u32)IN_ >> 16) & 0xFFFFu) | (32u << 16));                   // dim1 hi | tile_dim0=32
  g1[4] = rows;                                // tile_dim1 = rows (tile_dim2 = 0)
  g1[5] = IN_;                                 // tensor_dim0_stride[31:0]
  g1[6] = 0;                                   // stride0 hi | stride1 lo
  g1[7] = 0;
  v4i z4 = {0, 0, 0, 0};
  v8i z8 = {0, 0, 0, 0, 0, 0, 0, 0};
  __builtin_amdgcn_tensor_load_to_lds(g0, g1, z4, z4, z8, 0);
}
#else
#define HAVE_TDM 0
#endif

__device__ __forceinline__ void wait_tensorcnt_1() {
#if __has_builtin(__builtin_amdgcn_s_wait_tensorcnt)
  __builtin_amdgcn_s_wait_tensorcnt(1);
#else
  asm volatile("s_wait_tensorcnt 0x1" ::: "memory");
#endif
}
__device__ __forceinline__ void wait_tensorcnt_0() {
#if __has_builtin(__builtin_amdgcn_s_wait_tensorcnt)
  __builtin_amdgcn_s_wait_tensorcnt(0);
#else
  asm volatile("s_wait_tensorcnt 0x0" ::: "memory");
#endif
}

// ================= exact 2-rank radix select (12 + 10 + 10 bits) =================

__global__ __launch_bounds__(256) void hist_stage1(const float* __restrict__ W,
                                                   u32* __restrict__ hist) {
  __shared__ u32 lh[4096];
  for (int i = threadIdx.x; i < 4096; i += 256) lh[i] = 0;
  __syncthreads();
  const v4u* Wv = (const v4u*)W;
  const int n = NW / 4, stride = gridDim.x * blockDim.x;
  for (int i = blockIdx.x * blockDim.x + threadIdx.x; i < n; i += stride) {
    v4u w = Wv[i];
#pragma unroll
    for (int j = 0; j < 4; ++j) atomicAdd(&lh[fkey(w[j]) >> 20], 1u);
  }
  __syncthreads();
  for (int i = threadIdx.x; i < 4096; i += 256) {
    u32 c = lh[i];
    if (c) atomicAdd(&hist[i], c);
  }
}

__global__ __launch_bounds__(256) void hist_cond(const float* __restrict__ W,
                                                 const u32* __restrict__ ctrl,
                                                 u32* __restrict__ hist, int stage) {
  __shared__ u32 lh[2048];
  for (int i = threadIdx.x; i < 2048; i += 256) lh[i] = 0;
  __syncthreads();
  const u32 p0 = ctrl[0], p1 = ctrl[1];
  const int sh  = (stage == 2) ? 20 : 10;
  const int bsh = (stage == 2) ? 10 : 0;
  const v4u* Wv = (const v4u*)W;
  const int n = NW / 4, stride = gridDim.x * blockDim.x;
  for (int i = blockIdx.x * blockDim.x + threadIdx.x; i < n; i += stride) {
    v4u w = Wv[i];
#pragma unroll
    for (int j = 0; j < 4; ++j) {
      u32 k = fkey(w[j]);
      u32 top = k >> sh;
      u32 bin = (k >> bsh) & 1023u;
      if (top == p0) atomicAdd(&lh[bin], 1u);
      if (top == p1) atomicAdd(&lh[1024 + bin], 1u);
    }
  }
  __syncthreads();
  for (int i = threadIdx.x; i < 2048; i += 256) {
    u32 c = lh[i];
    if (c) atomicAdd(&hist[i], c);
  }
}

__global__ __launch_bounds__(1024) void select_scan(const u32* __restrict__ hist,
                                                    u32* __restrict__ ctrl,
                                                    int nbins, int stage) {
  __shared__ u32 part[1024];
  const int t = threadIdx.x;
  const int chunk = nbins / 1024;
  for (int j = 0; j < 2; ++j) {
    const u32* h  = (stage == 1) ? hist : hist + j * nbins;
    u32 rank = (stage == 1) ? (RANK0 + (u32)j) : ctrl[2 + j];
    u32 oldp = (stage == 1) ? 0u : ctrl[j];
    u32 local = 0;
    for (int i = 0; i < chunk; ++i) local += h[t * chunk + i];
    part[t] = local;
    __syncthreads();
    for (int off = 1; off < 1024; off <<= 1) {   // Hillis-Steele inclusive scan
      u32 v = (t >= off) ? part[t - off] : 0u;
      __syncthreads();
      part[t] += v;
      __syncthreads();
    }
    u32 cum = part[t] - local;                   // exclusive prefix of this chunk
    for (int i = 0; i < chunk; ++i) {
      u32 b = (u32)(t * chunk + i);
      u32 c = h[b];
      if (rank >= cum && rank < cum + c) {       // exactly one finder per target
        ctrl[j]     = (stage == 1) ? b : ((oldp << 10) | b);
        ctrl[2 + j] = rank - cum;
      }
      cum += c;
    }
    __syncthreads();
  }
}

__global__ void finalize_thr(u32* ctrl) {
  if (threadIdx.x == 0) {
    float f0 = keyval(ctrl[0]);
    float f1 = keyval(ctrl[1]);
    ((float*)ctrl)[6] = 0.5f * (f0 + f1);   // linear-interp quantile at q=0.5
  }
}

// ================= mask + split-bf16 conversion =================

__global__ __launch_bounds__(256) void convert_split_mask(const float* __restrict__ W,
                                                          const u32* __restrict__ ctrl,
                                                          u16* __restrict__ hi,
                                                          u16* __restrict__ lo) {
  const float thr = __uint_as_float(ctrl[6]);
  const v4u* Wv = (const v4u*)W;
  const int n = NW / 4, stride = gridDim.x * blockDim.x;
  for (int i = blockIdx.x * blockDim.x + threadIdx.x; i < n; i += stride) {
    v4u wb = Wv[i];
    u32 hp[2], lp[2];
#pragma unroll
    for (int j = 0; j < 4; ++j) {
      float w = __uint_as_float(wb[j]);
      float m = (w >= thr) ? w : 0.0f;
      u16 h = f2bf(m);
      u16 l = f2bf(m - bf2f(h));
      if (j & 1) { hp[j >> 1] |= ((u32)h) << 16; lp[j >> 1] |= ((u32)l) << 16; }
      else       { hp[j >> 1]  = h;              lp[j >> 1]  = l; }
    }
    v2u hv; hv[0] = hp[0]; hv[1] = hp[1];
    v2u lv; lv[0] = lp[0]; lv[1] = lp[1];
    *(v2u*)(hi + 4 * i) = hv;
    *(v2u*)(lo + 4 * i) = lv;
  }
}

__global__ __launch_bounds__(256) void convert_split(const float* __restrict__ X,
                                                     u16* __restrict__ hi,
                                                     u16* __restrict__ lo, int n4) {
  const v4u* Xv = (const v4u*)X;
  const int stride = gridDim.x * blockDim.x;
  for (int i = blockIdx.x * blockDim.x + threadIdx.x; i < n4; i += stride) {
    v4u wb = Xv[i];
    u32 hp[2], lp[2];
#pragma unroll
    for (int j = 0; j < 4; ++j) {
      float w = __uint_as_float(wb[j]);
      u16 h = f2bf(w);
      u16 l = f2bf(w - bf2f(h));
      if (j & 1) { hp[j >> 1] |= ((u32)h) << 16; lp[j >> 1] |= ((u32)l) << 16; }
      else       { hp[j >> 1]  = h;              lp[j >> 1]  = l; }
    }
    v2u hv; hv[0] = hp[0]; hv[1] = hp[1];
    v2u lv; lv[0] = lp[0]; lv[1] = lp[1];
    *(v2u*)(hi + 4 * i) = hv;
    *(v2u*)(lo + 4 * i) = lv;
  }
}

// ================= split-bf16 WMMA GEMM: out = X @ Wm^T + bias =================
// Block tile 64(M) x 128(N), k-tile 32, 8 waves (2x4), each wave 2x2 16x16 tiles.
// Tiles staged by the Tensor Data Mover (waves 0..3 own Xhi/Xlo/Whi/Wlo),
// double-buffered in LDS with TENSORcnt-based pipelining.

__global__ __launch_bounds__(256) void gemm_topkast(
    const u16* __restrict__ Xhi, const u16* __restrict__ Xlo,
    const u16* __restrict__ Whi, const u16* __restrict__ Wlo,
    const float* __restrict__ bias, float* __restrict__ out) {
  __shared__ u16 sXh[2][64][40];    // row stride 40 halfs = 80B (TDM pads 16B / 64B row)
  __shared__ u16 sXl[2][64][40];
  __shared__ u16 sWh[2][128][40];
  __shared__ u16 sWl[2][128][40];

  const int tid  = threadIdx.x;
  const int lane = tid & 31;
  const int wv   = tid >> 5;
  const int wm   = wv & 1;          // wave M index (0..1)
  const int wn   = wv >> 1;         // wave N index (0..3)
  const int half = (lane >> 4) & 1; // lane half per ISA fragment layout
  const int lm   = lane & 15;

  const int row0 = (blockIdx.x & 15) * 64;
  const int col0 = (blockIdx.x >> 4) * 128;

  const int xr = tid >> 2;          // fallback loader coords
  const int xc = (tid & 3) * 8;

  v8f acc[2][2] = {};
  union Frag { v16bf v; v4u q[2]; };

#if HAVE_TDM
  // each of waves 0..3 owns one operand stream
  const u16* src = (wv == 0) ? Xhi : (wv == 1) ? Xlo : (wv == 2) ? Whi : Wlo;
  const int rows = (wv < 2) ? 64 : 128;
  const int rb   = (wv < 2) ? row0 : col0;
  u32 lds0 = 0, lds1 = 0;
  if (wv == 0)      { lds0 = (u32)(uintptr_t)&sXh[0][0][0]; lds1 = (u32)(uintptr_t)&sXh[1][0][0]; }
  else if (wv == 1) { lds0 = (u32)(uintptr_t)&sXl[0][0][0]; lds1 = (u32)(uintptr_t)&sXl[1][0][0]; }
  else if (wv == 2) { lds0 = (u32)(uintptr_t)&sWh[0][0][0]; lds1 = (u32)(uintptr_t)&sWh[1][0][0]; }
  else if (wv == 3) { lds0 = (u32)(uintptr_t)&sWl[0][0][0]; lds1 = (u32)(uintptr_t)&sWl[1][0][0]; }

  if (wv < 4) tdm_load_tile(lds0, src + rb * IN_, rows);   // stage 0 -> buf 0
#endif

  for (int kt = 0; kt < 128; ++kt) {
    const int buf = kt & 1;
#if HAVE_TDM
    __syncthreads();   // prior readers of buf^1 are done before TDM overwrites it
    if (wv < 4) {
      if (kt + 1 < 128) {
        tdm_load_tile((kt & 1) ? lds0 : lds1, src + rb * IN_ + (kt + 1) * 32, rows);
        wait_tensorcnt_1();   // TDM in-order per wave: stage kt complete
      } else {
        wait_tensorcnt_0();
      }
    }
    __syncthreads();   // stage kt visible to all waves
#else
    const int k0 = kt * 32;
    __syncthreads();
    *(v4u*)&sXh[0][xr][xc] = *(const v4u*)(Xhi + (row0 + xr) * IN_ + k0 + xc);
    *(v4u*)&sXl[0][xr][xc] = *(const v4u*)(Xlo + (row0 + xr) * IN_ + k0 + xc);
#pragma unroll
    for (int i = 0; i < 2; ++i) {
      const int c  = tid + i * 256;
      const int wr = c >> 2;
      const int wc = (c & 3) * 8;
      *(v4u*)&sWh[0][wr][wc] = *(const v4u*)(Whi + (col0 + wr) * IN_ + k0 + wc);
      *(v4u*)&sWl[0][wr][wc] = *(const v4u*)(Wlo + (col0 + wr) * IN_ + k0 + wc);
    }
    if (kt + 1 < 128) {
      __builtin_prefetch(Whi + (col0 + 2 * xr) * IN_ + k0 + 32, 0, 1);
      __builtin_prefetch(Xhi + (row0 + xr) * IN_ + k0 + 32, 0, 1);
    }
    __syncthreads();
#endif

    const u16 (*pXh)[40] = sXh[HAVE_TDM ? buf : 0];
    const u16 (*pXl)[40] = sXl[HAVE_TDM ? buf : 0];
    const u16 (*pWh)[40] = sWh[HAVE_TDM ? buf : 0];
    const u16 (*pWl)[40] = sWl[HAVE_TDM ? buf : 0];

    Frag ah[2], al[2], bh[2], bl[2];
#pragma unroll
    for (int mi = 0; mi < 2; ++mi) {   // A 16x32 bf16: K chunks [8h..8h+7], [16+8h..]
      const int r = wm * 32 + mi * 16 + lm;
      ah[mi].q[0] = *(const v4u*)&pXh[r][half * 8];
      ah[mi].q[1] = *(const v4u*)&pXh[r][16 + half * 8];
      al[mi].q[0] = *(const v4u*)&pXl[r][half * 8];
      al[mi].q[1] = *(const v4u*)&pXl[r][16 + half * 8];
    }
#pragma unroll
    for (int ni = 0; ni < 2; ++ni) {   // B 32x16 bf16: col=lm, K chunks [8h..8h+7], [16+8h..]
      const int r = wn * 32 + ni * 16 + lm;
      bh[ni].q[0] = *(const v4u*)&pWh[r][half * 8];
      bh[ni].q[1] = *(const v4u*)&pWh[r][16 + half * 8];
      bl[ni].q[0] = *(const v4u*)&pWl[r][half * 8];
      bl[ni].q[1] = *(const v4u*)&pWl[r][16 + half * 8];
    }
#pragma unroll
    for (int mi = 0; mi < 2; ++mi)
#pragma unroll
      for (int ni = 0; ni < 2; ++ni) {
        acc[mi][ni] = __builtin_amdgcn_wmma_f32_16x16x32_bf16(
            false, ah[mi].v, false, bh[ni].v, (short)0, acc[mi][ni], false, false);
        acc[mi][ni] = __builtin_amdgcn_wmma_f32_16x16x32_bf16(
            false, ah[mi].v, false, bl[ni].v, (short)0, acc[mi][ni], false, false);
        acc[mi][ni] = __builtin_amdgcn_wmma_f32_16x16x32_bf16(
            false, al[mi].v, false, bh[ni].v, (short)0, acc[mi][ni], false, false);
      }
  }

  // epilogue: C/D layout -> VGPR r holds M = r + 8*half, N = lane&15
#pragma unroll
  for (int mi = 0; mi < 2; ++mi)
#pragma unroll
    for (int ni = 0; ni < 2; ++ni) {
      const int r0 = row0 + wm * 32 + mi * 16 + half * 8;
      const int cc = col0 + wn * 32 + ni * 16 + lm;
      const float bb = bias[cc];
#pragma unroll
      for (int r = 0; r < 8; ++r)
        out[(r0 + r) * OUT_ + cc] = acc[mi][ni][r] + bb;
    }
}

// ================= host launch =================

extern "C" void kernel_launch(void* const* d_in, const int* in_sizes, int n_in,
                              void* d_out, int out_size, void* d_ws, size_t ws_size,
                              hipStream_t stream) {
  const float* X    = (const float*)d_in[0];   // [1024, 4096]
  const float* W    = (const float*)d_in[1];   // [4096, 4096]
  const float* bias = (const float*)d_in[2];   // [4096]
  float* out = (float*)d_out;                  // [1024, 4096]
  char* ws = (char*)d_ws;

  u32* hist1 = (u32*)(ws);            // 4096 bins
  u32* hist2 = (u32*)(ws + 16384);    // 2 x 1024 bins
  u32* hist3 = (u32*)(ws + 24576);    // 2 x 1024 bins
  u32* ctrl  = (u32*)(ws + 32768);    // prefixes/ranks/threshold
  u16* Whi = (u16*)(ws + 65536);
  u16* Wlo = Whi + (size_t)NW;
  u16* Xhi = Wlo + (size_t)NW;
  u16* Xlo = Xhi + (size_t)(B_ * IN_);

  (void)hipMemsetAsync(d_ws, 0, 32768 + 64, stream);

  // exact median via 3-pass radix select on order-preserving keys
  hist_stage1<<<256, 256, 0, stream>>>(W, hist1);
  select_scan<<<1, 1024, 0, stream>>>(hist1, ctrl, 4096, 1);
  hist_cond<<<256, 256, 0, stream>>>(W, ctrl, hist2, 2);
  select_scan<<<1, 1024, 0, stream>>>(hist2, ctrl, 1024, 2);
  hist_cond<<<256, 256, 0, stream>>>(W, ctrl, hist3, 3);
  select_scan<<<1, 1024, 0, stream>>>(hist3, ctrl, 1024, 3);
  finalize_thr<<<1, 32, 0, stream>>>(ctrl);

  // mask + split into bf16 hi/lo (fits in 192MB L2 for the GEMM)
  convert_split_mask<<<2048, 256, 0, stream>>>(W, ctrl, Whi, Wlo);
  convert_split<<<1024, 256, 0, stream>>>(X, Xhi, Xlo, (B_ * IN_) / 4);

  // 3-term split-bf16 WMMA GEMM, TDM-staged tiles, fp32 accumulation
  gemm_topkast<<<512, 256, 0, stream>>>(Xhi, Xlo, Whi, Wlo, bias, out);
}
